// _FallbackEdgeGraphSAGE_6519760355644
// MI455X (gfx1250) — matrix operand order repaired and checked by reading, not compiled, not run on hardware
//
#include <hip/hip_runtime.h>
#include <hip/hip_bf16.h>
#include <hip/hip_fp16.h>

// ---------------------------------------------------------------------------
// EdgeGraphSAGE for MI455X (gfx1250): wave32 + v_wmma_f32_16x16x32_f16
// Round 4: revert sched_group_barrier (it caused accumulator spills to
// scratch). Rely on multi-wave occupancy to hide L1 B-fragment latency —
// round-2 codegen was spill-free with 124 WMMA sites.
// ---------------------------------------------------------------------------

typedef __attribute__((ext_vector_type(16))) _Float16 v16h;
typedef __attribute__((ext_vector_type(8)))  float    v8f;

#define NN   100000
#define EE   1600000
#define DIN  64
#define HH   128
#define EIN  64
#define CCC  8
#define EPSV 1e-5f

#define WMMA16(a, b, c) \
  __builtin_amdgcn_wmma_f32_16x16x32_f16(false, (a), false, (b), (short)0, (c), false, false)

// ---- fragment builders ----------------------------------------------------
// A-matrix (16-bit, 16x32, MxK): lane&15 = M for both halves; lane>>4 selects
// K subset: VGPR0..3 = K in [kbase+half*8, +8), VGPR4..7 = K in [kbase+16+half*8, +8)
static __device__ __forceinline__ v16h afrag_f16(const _Float16* row, int kbase, int half) {
  union { v16h v; float4 q[2]; } u;
  u.q[0] = *reinterpret_cast<const float4*>(row + kbase + half * 8);
  u.q[1] = *reinterpret_cast<const float4*>(row + kbase + 16 + half * 8);
  return u.v;
}
static __device__ __forceinline__ v16h afrag_f32(const float* row, int kbase, int half) {
  union { v16h v; _Float16 h[16]; } u;
  const float* p0 = row + kbase + half * 8;
  const float* p1 = row + kbase + 16 + half * 8;
#pragma unroll
  for (int i = 0; i < 8; ++i) {
    u.h[i]     = (_Float16)p0[i];
    u.h[8 + i] = (_Float16)p1[i];
  }
  return u.v;
}
// B-matrix (16-bit, 32x16, KxN): lane l holds column N = n0 + (l&15); K = kblock*32 + half*8 + i, then +16
static __device__ __forceinline__ v16h bfrag(const _Float16* B, int ldb, int krow, int n0) {
  union { v16h v; _Float16 h[16]; } u;
  int l = krow & 31, kb0 = krow - l, hf = l >> 4;
  const _Float16* p = B + (size_t)(kb0 + hf * 8) * ldb + n0 + (l & 15);
#pragma unroll
  for (int i = 0; i < 8; ++i) {
    u.h[i]     = p[(size_t)i * ldb];
    u.h[8 + i] = p[(size_t)(16 + i) * ldb];
  }
  return u.v;
}

// ---- utility kernels ------------------------------------------------------
__global__ void k_zero(float* p, long n) {
  for (long i = (long)blockIdx.x * blockDim.x + threadIdx.x; i < n;
       i += (long)gridDim.x * blockDim.x) p[i] = 0.0f;
}

__global__ void k_deg(const int* dst, float* deg) {
  for (long e = (long)blockIdx.x * blockDim.x + threadIdx.x; e < EE;
       e += (long)gridDim.x * blockDim.x) atomicAdd(&deg[dst[e]], 1.0f);
}

__global__ void k_invdeg(const float* deg, float* inv) {
  for (long i = (long)blockIdx.x * blockDim.x + threadIdx.x; i < NN;
       i += (long)gridDim.x * blockDim.x) {
    float d = deg[i];
    inv[i] = d > 0.0f ? 1.0f / d : 0.0f;
  }
}

__global__ void k_cvt16(_Float16* d, const float* s, long n) {
  for (long i = (long)blockIdx.x * blockDim.x + threadIdx.x; i < n;
       i += (long)gridDim.x * blockDim.x) d[i] = (_Float16)s[i];
}

// dst[(rowsA+rowsB) x cols] f16 = [A; B]
__global__ void k_cat2(_Float16* d, const float* a, const float* b, int rowsA, int rowsT, int cols) {
  long n = (long)rowsT * cols;
  for (long i = (long)blockIdx.x * blockDim.x + threadIdx.x; i < n;
       i += (long)gridDim.x * blockDim.x) {
    int r = (int)(i / cols), c = (int)(i - (long)r * cols);
    float v = (r < rowsA) ? a[(long)r * cols + c] : b[(long)(r - rowsA) * cols + c];
    d[i] = (_Float16)v;
  }
}

// We2 [128 x 8] f32 -> [128 x 16] f16 zero-padded
__global__ void k_padwe2(_Float16* d, const float* s) {
  for (long i = (long)blockIdx.x * blockDim.x + threadIdx.x; i < 128L * 16;
       i += (long)gridDim.x * blockDim.x) {
    int r = (int)(i >> 4), c = (int)(i & 15);
    d[i] = (c < CCC) ? (_Float16)s[r * CCC + c] : (_Float16)0.0f;
  }
}

// xa[:, 0:64] = f16(x)  (row stride 128)
__global__ void k_packx(_Float16* xa, const float* x) {
  long n = (long)NN * DIN;
  for (long i = (long)blockIdx.x * blockDim.x + threadIdx.x; i < n;
       i += (long)gridDim.x * blockDim.x) {
    int row = (int)(i / DIN), j = (int)(i - (long)row * DIN);
    xa[(size_t)row * 128 + j] = (_Float16)x[i];
  }
}

// xa[:, off:off+w] = f16(agg * inv_deg)
__global__ void k_packagg(_Float16* xa, int stride, int off, const float* agg, int w,
                          const float* inv) {
  long n = (long)NN * w;
  for (long i = (long)blockIdx.x * blockDim.x + threadIdx.x; i < n;
       i += (long)gridDim.x * blockDim.x) {
    int row = (int)(i / w), j = (int)(i - (long)row * w);
    xa[(size_t)row * stride + off + j] = (_Float16)(agg[i] * inv[row]);
  }
}

// agg[dst[e], :] += feat[src[e], :]
__global__ void k_scatter(float* agg, const int* src, const int* dst, const float* feat, int w) {
  long n = (long)EE * w;
  for (long i = (long)blockIdx.x * blockDim.x + threadIdx.x; i < n;
       i += (long)gridDim.x * blockDim.x) {
    int e = (int)(i / w), j = (int)(i - (long)e * w);
    atomicAdd(&agg[(size_t)dst[e] * w + j], feat[(size_t)src[e] * w + j]);
  }
}

// ---- node GEMM: z[N,128] = A_f16[N,K] @ B_f16[K,128] + bias; BN partial stats
__global__ void k_gemm_node(const _Float16* __restrict__ A, int lda, int kblocks,
                            const _Float16* __restrict__ B, const float* __restrict__ bias,
                            float* __restrict__ z, float* __restrict__ cs,
                            float* __restrict__ cq, int ntiles) {
  int wave = threadIdx.x >> 5;
  int lane = threadIdx.x & 31;
  int tile = blockIdx.x * (blockDim.x >> 5) + wave;
  if (tile >= ntiles) return;  // wave-uniform: EXEC stays all-ones for WMMA
  int m16 = lane & 15, half = lane >> 4;
  const _Float16* arow = A + ((size_t)tile * 16 + m16) * lda;

  v8f cc[8] = {};
  for (int kb = 0; kb < kblocks; ++kb) {
    v16h a = afrag_f16(arow, kb * 32, half);
#pragma unroll
    for (int nt = 0; nt < 8; ++nt) {
      v16h b = bfrag(B, 128, kb * 32 + lane, nt * 16);
      cc[nt] = WMMA16(a, b, cc[nt]);
    }
  }
#pragma unroll
  for (int nt = 0; nt < 8; ++nt) {
    int n = nt * 16 + m16;
    float bn = bias[n];
    float s = 0.0f, q = 0.0f;
#pragma unroll
    for (int r = 0; r < 8; ++r) {
      int m = half * 8 + r;
      float val = cc[nt][r] + bn;
      z[((size_t)tile * 16 + m) * 128 + n] = val;
      s += val;
      q += val * val;
    }
    s += __shfl_xor(s, 16, 32);
    q += __shfl_xor(q, 16, 32);
    if (lane < 16) {
      atomicAdd(&cs[n], s);
      atomicAdd(&cq[n], q);
    }
  }
}

__global__ void k_bnstat(const float* cs, const float* cq, float* meanr, float* rstd) {
  int n = threadIdx.x;
  if (n < HH) {
    float m = cs[n] / (float)NN;
    float v = cq[n] / (float)NN - m * m;
    meanr[n] = m;
    rstd[n] = rsqrtf(v + EPSV);
  }
}

// h = relu((z-mean)*rstd*g + beta); optional f32 and f16 outputs
__global__ void k_bnapply(const float* __restrict__ z, const float* __restrict__ meanr,
                          const float* __restrict__ rstd, const float* __restrict__ g,
                          const float* __restrict__ bt, float* h32, _Float16* h16,
                          int h16stride) {
  long n = (long)NN * HH;
  for (long i = (long)blockIdx.x * blockDim.x + threadIdx.x; i < n;
       i += (long)gridDim.x * blockDim.x) {
    int row = (int)(i / HH), j = (int)(i - (long)row * HH);
    float val = (z[i] - meanr[j]) * rstd[j] * g[j] + bt[j];
    val = fmaxf(val, 0.0f);
    if (h32) h32[i] = val;
    if (h16) h16[(size_t)row * h16stride + j] = (_Float16)val;
  }
}

// ---- edge head: per-wave 16-edge tile, two chained WMMA GEMMs -------------
__global__ void __launch_bounds__(256) k_edge(const _Float16* __restrict__ h2,
                                              const float* __restrict__ ef,
                                              const int* __restrict__ u,
                                              const int* __restrict__ v,
                                              const _Float16* __restrict__ W1h,
                                              const float* __restrict__ be1,
                                              const _Float16* __restrict__ W2p,
                                              const float* __restrict__ be2,
                                              float* __restrict__ out) {
  __shared__ __align__(16) _Float16 hidbuf[8][16 * 136];
  int wave = threadIdx.x >> 5;
  int lane = threadIdx.x & 31;
  int tile = blockIdx.x * 8 + wave;
  if (tile >= EE / 16) return;  // wave-uniform
  int e0 = tile * 16;
  int m16 = lane & 15, half = lane >> 4;

  const _Float16* pu = h2 + (size_t)u[e0 + m16] * HH;
  const _Float16* pv = h2 + (size_t)v[e0 + m16] * HH;
  const float*    pe = ef + (size_t)(e0 + m16) * EIN;

  // GEMM1: [16 x 320] @ We1 -> hid [16 x 128], K-blocks: 4 from h[u], 4 from h[v], 2 from e_feat
  v8f cc[8] = {};
  for (int kb = 0; kb < 10; ++kb) {
    v16h a;
    if (kb < 4)      a = afrag_f16(pu, kb * 32, half);
    else if (kb < 8) a = afrag_f16(pv, (kb - 4) * 32, half);
    else             a = afrag_f32(pe, (kb - 8) * 32, half);  // on-the-fly f32->f16
#pragma unroll
    for (int nt = 0; nt < 8; ++nt) {
      v16h b = bfrag(W1h, 128, kb * 32 + lane, nt * 16);
      cc[nt] = WMMA16(a, b, cc[nt]);
    }
  }

  // bias + ReLU -> f16 LDS tile (C-layout -> A-layout reshaping buffer)
  _Float16* hid = &hidbuf[wave][0];
#pragma unroll
  for (int nt = 0; nt < 8; ++nt) {
    int n = nt * 16 + m16;
    float bn = be1[n];
#pragma unroll
    for (int r = 0; r < 8; ++r) {
      int m = half * 8 + r;
      hid[m * 136 + n] = (_Float16)fmaxf(cc[nt][r] + bn, 0.0f);
    }
  }

  // GEMM2: hid [16 x 128] @ We2pad [128 x 16] (cols 8..15 zero)
  v8f oc = {};
  const _Float16* hrow = hid + m16 * 136;
#pragma unroll
  for (int kb = 0; kb < 4; ++kb) {
    union { v16h v; float4 q[2]; } ua;
    ua.q[0] = *reinterpret_cast<const float4*>(hrow + kb * 32 + half * 8);
    ua.q[1] = *reinterpret_cast<const float4*>(hrow + kb * 32 + 16 + half * 8);
    v16h b = bfrag(W2p, 16, kb * 32 + lane, 0);
    oc = WMMA16(ua.v, b, oc);
  }

  int n = m16;
  if (n < CCC) {
    float bb = be2[n];
#pragma unroll
    for (int r = 0; r < 8; ++r) {
      int m = half * 8 + r;
      out[(size_t)(e0 + m) * CCC + n] = oc[r] + bb;
    }
  }
}

// ---------------------------------------------------------------------------
extern "C" void kernel_launch(void* const* d_in, const int* in_sizes, int n_in,
                              void* d_out, int out_size, void* d_ws, size_t ws_size,
                              hipStream_t stream) {
  (void)in_sizes; (void)n_in; (void)out_size; (void)ws_size;
  const float* x   = (const float*)d_in[0];
  const float* ef  = (const float*)d_in[1];
  const float* Ws1 = (const float*)d_in[2];
  const float* Wn1 = (const float*)d_in[3];
  const float* b1  = (const float*)d_in[4];
  const float* g1  = (const float*)d_in[5];
  const float* bt1 = (const float*)d_in[6];
  const float* Ws2 = (const float*)d_in[7];
  const float* Wn2 = (const float*)d_in[8];
  const float* b2  = (const float*)d_in[9];
  const float* g2  = (const float*)d_in[10];
  const float* bt2 = (const float*)d_in[11];
  const float* We1 = (const float*)d_in[12];
  const float* be1 = (const float*)d_in[13];
  const float* We2 = (const float*)d_in[14];
  const float* be2 = (const float*)d_in[15];
  const int* srcI  = (const int*)d_in[16];
  const int* dstI  = (const int*)d_in[17];
  const int* uI    = (const int*)d_in[18];
  const int* vI    = (const int*)d_in[19];
  float* out = (float*)d_out;

  char* w = (char*)d_ws;
  size_t off = 0;
  auto alloc = [&](size_t bytes) -> char* {
    char* p = w + off;
    off = (off + bytes + 255) & ~(size_t)255;
    return p;
  };
  float*     deg   = (float*)alloc((size_t)NN * 4);
  float*     inv   = (float*)alloc((size_t)NN * 4);
  float*     cs    = (float*)alloc(HH * 4);
  float*     cq    = (float*)alloc(HH * 4);
  float*     meanr = (float*)alloc(HH * 4);
  float*     rstd  = (float*)alloc(HH * 4);
  _Float16*  Wc1   = (_Float16*)alloc(128 * 128 * 2);
  _Float16*  Wc2   = (_Float16*)alloc(256 * 128 * 2);
  _Float16*  W1h   = (_Float16*)alloc(320 * 128 * 2);
  _Float16*  W2p   = (_Float16*)alloc(128 * 16 * 2);
  _Float16*  xa    = (_Float16*)alloc((size_t)NN * 256 * 2);
  float*     agg   = (float*)alloc((size_t)NN * 128 * 4);
  float*     zbuf  = (float*)alloc((size_t)NN * 128 * 4);
  float*     h1    = (float*)alloc((size_t)NN * 128 * 4);
  _Float16*  h2    = (_Float16*)alloc((size_t)NN * 128 * 2);

  const int T = 256;
  auto nb = [](long n, int t) -> int {
    long b = (n + t - 1) / t;
    if (b > 1048576) b = 1048576;
    return (int)(b > 0 ? b : 1);
  };

  // ---- setup: degree, weights, x pack
  k_zero<<<nb(NN, T), T, 0, stream>>>(deg, NN);
  k_zero<<<1, 256, 0, stream>>>(cs, HH);
  k_zero<<<1, 256, 0, stream>>>(cq, HH);
  k_zero<<<nb((long)NN * 64, T), T, 0, stream>>>(agg, (long)NN * 64);
  k_deg<<<nb(EE, T), T, 0, stream>>>(dstI, deg);
  k_invdeg<<<nb(NN, T), T, 0, stream>>>(deg, inv);
  k_cvt16<<<nb(320L * 128, T), T, 0, stream>>>(W1h, We1, 320L * 128);
  k_cat2<<<nb(128L * 128, T), T, 0, stream>>>(Wc1, Ws1, Wn1, 64, 128, 128);
  k_cat2<<<nb(256L * 128, T), T, 0, stream>>>(Wc2, Ws2, Wn2, 128, 256, 128);
  k_padwe2<<<nb(128L * 16, T), T, 0, stream>>>(W2p, We2);
  k_packx<<<nb((long)NN * DIN, T), T, 0, stream>>>(xa, x);

  // ---- SAGE layer 1
  k_scatter<<<nb((long)EE * 64, T), T, 0, stream>>>(agg, srcI, dstI, x, 64);
  k_packagg<<<nb((long)NN * 64, T), T, 0, stream>>>(xa, 128, 64, agg, 64, inv);
  k_gemm_node<<<(6250 + 3) / 4, 128, 0, stream>>>(xa, 128, 4, Wc1, b1, zbuf, cs, cq, 6250);
  k_bnstat<<<1, 128, 0, stream>>>(cs, cq, meanr, rstd);
  k_bnapply<<<nb((long)NN * HH, T), T, 0, stream>>>(zbuf, meanr, rstd, g1, bt1, h1, xa, 256);

  // ---- SAGE layer 2
  k_zero<<<nb((long)NN * 128, T), T, 0, stream>>>(agg, (long)NN * 128);
  k_zero<<<1, 256, 0, stream>>>(cs, HH);
  k_zero<<<1, 256, 0, stream>>>(cq, HH);
  k_scatter<<<nb((long)EE * 128, T), T, 0, stream>>>(agg, srcI, dstI, h1, 128);
  k_packagg<<<nb((long)NN * 128, T), T, 0, stream>>>(xa, 256, 128, agg, 128, inv);
  k_gemm_node<<<(6250 + 3) / 4, 128, 0, stream>>>(xa, 256, 8, Wc2, b2, zbuf, cs, cq, 6250);
  k_bnstat<<<1, 128, 0, stream>>>(cs, cq, meanr, rstd);
  k_bnapply<<<nb((long)NN * HH, T), T, 0, stream>>>(zbuf, meanr, rstd, g2, bt2, nullptr, h2, 128);

  // ---- edge head: 100000 tiles of 16 edges, 8 waves/block
  k_edge<<<12500, 256, 0, stream>>>(h2, ef, uI, vI, W1h, be1, W2p, be2, out);
}
